// Flashback_4209067950804
// MI455X (gfx1250) — compile-verified, matrix operand order checked by reading, not run on hardware
//
#include <hip/hip_runtime.h>
#include <hip/hip_bf16.h>
#include <math.h>

// ---------------------------------------------------------------------------
// Sizes from the reference
// ---------------------------------------------------------------------------
static constexpr int S = 64, U = 256, H = 128, H2 = 256;
static constexpr int NLOC = 10000;
static constexpr int M = S * U;            // 16384 rows of the final GEMM
static constexpr int MT = M / 16;          // 1024 m-tiles
static constexpr int NSTRIPS = 25;         // 625 n-tiles = 25 strips * 25 tiles
static constexpr int NT_PER_STRIP = 25;
static constexpr int MG = MT / 8;          // 128 m-groups (8 m-tiles per WG)

typedef _Float16 v8h  __attribute__((ext_vector_type(8)));
typedef _Float16 v16h __attribute__((ext_vector_type(16)));
typedef float    v8f  __attribute__((ext_vector_type(8)));

// workspace layout (bytes)
static constexpr size_t OFF_XEMB = 0;                               // S*U*H f32  = 8 MB
static constexpr size_t OFF_OUT  = OFF_XEMB + (size_t)S*U*H*4;      // S*U*H f32  = 8 MB
static constexpr size_t OFF_PU   = OFF_OUT  + (size_t)S*U*H*4;      // U*H f32
static constexpr size_t OFF_SIM  = OFF_PU   + (size_t)U*H*4;        // U*S f32
static constexpr size_t OFF_COEF = OFF_SIM  + (size_t)U*S*4;        // S*U f32
static constexpr size_t OFF_A    = OFF_COEF + (size_t)S*U*4;        // M*2H f16   = 8 MB
static constexpr size_t OFF_WH   = OFF_A    + (size_t)M*H2*2;       // NLOC*2H f16= 5 MB

// ---------------------------------------------------------------------------
// 1) gathers
// ---------------------------------------------------------------------------
__global__ void k_gather_x(const int* __restrict__ x, const float* __restrict__ enc,
                           float* __restrict__ xemb) {
    int idx = blockIdx.x * blockDim.x + threadIdx.x;       // S*U*H
    if (idx >= S * U * H) return;
    int row = idx >> 7, k = idx & 127;
    xemb[idx] = enc[(size_t)x[row] * H + k];
}

__global__ void k_gather_u(const int* __restrict__ au, const float* __restrict__ uemb,
                           float* __restrict__ pu) {
    int idx = blockIdx.x * blockDim.x + threadIdx.x;       // U*H
    if (idx >= U * H) return;
    int u = idx >> 7, k = idx & 127;
    pu[idx] = uemb[(size_t)au[u] * H + k];
}

// ---------------------------------------------------------------------------
// 2) tanh RNN: one block per user, 128 threads (one per hidden unit)
// ---------------------------------------------------------------------------
__global__ void __launch_bounds__(128) k_rnn(const float* __restrict__ xemb,
                                             const float* __restrict__ h0,
                                             const float* __restrict__ W_ih,
                                             const float* __restrict__ b_ih,
                                             const float* __restrict__ W_hh,
                                             const float* __restrict__ b_hh,
                                             float* __restrict__ out,
                                             float* __restrict__ hout) {
    int u = blockIdx.x, k = threadIdx.x;
    __shared__ float hp[H];
    __shared__ float xr[H];
    hp[k] = h0[u * H + k];
    float bsum = b_ih[k] + b_hh[k];
    const float* wi = W_ih + (size_t)k * H;
    const float* wh = W_hh + (size_t)k * H;
    for (int st = 0; st < S; ++st) {
        xr[k] = xemb[((size_t)st * U + u) * H + k];
        __syncthreads();
        float acc = bsum;
        #pragma unroll 4
        for (int j = 0; j < H; ++j) acc += xr[j] * wi[j] + hp[j] * wh[j];
        float hn = tanhf(acc);
        __syncthreads();
        hp[k] = hn;
        out[((size_t)st * U + u) * H + k] = hn;
        __syncthreads();
    }
    hout[u * H + k] = hp[k];
}

// ---------------------------------------------------------------------------
// 3) cosine similarity: one block per user, 256 threads (one per 2H dim)
// ---------------------------------------------------------------------------
__global__ void __launch_bounds__(256) k_sim(const float* __restrict__ pu,
                                             const float* __restrict__ xemb,
                                             const float* __restrict__ Wp,
                                             const float* __restrict__ bp,
                                             const float* __restrict__ pref,
                                             float* __restrict__ sim) {
    int u = blockIdx.x, k = threadIdx.x;                   // k in [0,256)
    __shared__ float pur[H];
    __shared__ float xr[H];
    __shared__ float red1[H2];
    __shared__ float red2[H2];
    if (k < H) pur[k] = pu[u * H + k];
    __syncthreads();
    const float* wrow = Wp + (size_t)k * H;
    float acc = bp[k];
    #pragma unroll 4
    for (int j = 0; j < H; ++j) acc += pur[j] * wrow[j];
    float v1 = tanhf(acc) + pref[k];
    red1[k] = v1 * v1;
    __syncthreads();
    for (int st = H; st > 0; st >>= 1) {
        if (k < st) red1[k] += red1[k + st];
        __syncthreads();
    }
    float n1 = sqrtf(red1[0]);
    for (int sstep = 0; sstep < S; ++sstep) {
        __syncthreads();
        if (k < H) xr[k] = xemb[((size_t)sstep * U + u) * H + k];
        __syncthreads();
        float a2 = bp[k];
        #pragma unroll 4
        for (int j = 0; j < H; ++j) a2 += xr[j] * wrow[j];
        float v2 = tanhf(a2);
        red1[k] = v1 * v2;
        red2[k] = v2 * v2;
        __syncthreads();
        for (int st = H; st > 0; st >>= 1) {
            if (k < st) { red1[k] += red1[k + st]; red2[k] += red2[k + st]; }
            __syncthreads();
        }
        if (k == 0) {
            float den = fmaxf(n1 * sqrtf(red2[0]), 1e-8f);
            sim[u * S + sstep] = 0.5f * (1.0f + red1[0] / den);
        }
    }
}

// ---------------------------------------------------------------------------
// 4) causal decay weights -> coef[i,u] = w_ii / sum_{j<=i} w_ij
// ---------------------------------------------------------------------------
__global__ void k_coef(const float* __restrict__ t, const float* __restrict__ sp,
                       const float* __restrict__ sim, float* __restrict__ coef) {
    int idx = blockIdx.x * blockDim.x + threadIdx.x;       // S*U
    if (idx >= S * U) return;
    int i = idx / U, u = idx % U;
    float ti = t[i * U + u];
    float sx = sp[(size_t)(i * U + u) * 2 + 0];
    float sy = sp[(size_t)(i * U + u) * 2 + 1];
    float simf = sim[u * S + i];
    const float W2PI = 7.2722052166430399e-05f;  // 2*pi/86400
    const float LT   = 1.1574074074074074e-06f;  // 0.1/86400
    float sum = 0.0f;
    for (int j = 0; j <= i; ++j) {
        float dt  = ti - t[j * U + u];
        float dx  = sx - sp[(size_t)(j * U + u) * 2 + 0];
        float dy  = sy - sp[(size_t)(j * U + u) * 2 + 1];
        float ds  = sqrtf(dx * dx + dy * dy + 1e-12f);
        float ft  = (cosf(dt * W2PI) + 1.0f) * 0.5f * expf(-dt * LT);
        float fs  = expf(-ds * 100.0f);
        sum += (ft * fs + 1e-10f) * simf;
    }
    float wii = (expf(-sqrtf(1e-12f) * 100.0f) + 1e-10f) * simf;  // ft(0)==1
    coef[idx] = wii / sum;
}

// ---------------------------------------------------------------------------
// 5) pack A = [coef*out | p_u] and W_fc into f16 for WMMA
// ---------------------------------------------------------------------------
__global__ void k_pack_a(const float* __restrict__ outr, const float* __restrict__ coef,
                         const float* __restrict__ pu, _Float16* __restrict__ A) {
    int idx = blockIdx.x * blockDim.x + threadIdx.x;       // M*2H
    if (idx >= M * H2) return;
    int row = idx >> 8, k2 = idx & 255;
    int u = row & (U - 1);
    float v = (k2 < H) ? coef[row] * outr[(size_t)row * H + k2]
                       : pu[u * H + (k2 - H)];
    A[idx] = (_Float16)v;
}

__global__ void k_pack_w(const float* __restrict__ W, _Float16* __restrict__ Wh) {
    int idx = blockIdx.x * blockDim.x + threadIdx.x;       // NLOC*2H
    if (idx >= NLOC * H2) return;
    Wh[idx] = (_Float16)W[idx];
}

// ---------------------------------------------------------------------------
// 6) WMMA GEMM: Y[M,NLOC] = A[M,256] * Wh[NLOC,256]^T + b_fc
//
//    Workgroup = 8 waves = 8 m-tiles; walks a 25-tile N-strip.
//    Each 8KB W_fc tile (16 rows x 256 f16, contiguous) is staged ONCE in
//    LDS (double-buffered) and shared by all 8 waves -> 8x less B traffic,
//    bringing B reads (~640 MB) to parity with the NT store stream (655 MB).
//    A fragments stay register-resident per wave. One barrier per tile:
//    buffer b is rewritten two iterations later, after the next barrier.
// ---------------------------------------------------------------------------
__global__ void __launch_bounds__(256) k_gemm(const _Float16* __restrict__ A,
                                              const _Float16* __restrict__ B,
                                              const float* __restrict__ bias,
                                              float* __restrict__ Y) {
    __shared__ alignas(16) _Float16 bsh[2][16 * H2];   // 2 x 8 KB

    int tid  = threadIdx.x;
    int wave = tid >> 5;
    int lane = tid & 31;
    int mgroup = blockIdx.x / NSTRIPS;   // 0..127
    int nstrip = blockIdx.x % NSTRIPS;   // 0..24
    int mtile  = mgroup * 8 + wave;      // 0..1023
    int m16 = lane & 15;
    int hs  = lane >> 4;                 // which K-half this half-wave holds

    // A fragments: lane m16 holds row (mtile*16+m16); per 32-K chunk the lane
    // holds K = hs*8 + [0..7] and 16 + hs*8 + [0..7]  -> two b128 loads.
    v16h a[8];
    const _Float16* arow = A + ((size_t)(mtile * 16 + m16)) * H2 + hs * 8;
    #pragma unroll
    for (int kc = 0; kc < 8; ++kc) {
        union { v16h v; v8h h[2]; } ua;
        ua.h[0] = *(const v8h*)(arow + kc * 32);
        ua.h[1] = *(const v8h*)(arow + kc * 32 + 16);
        a[kc] = ua.v;
    }

    // strip base: n-tile rows are contiguous in Wh -> each tile is one 8KB block
    const _Float16* bstrip = B + (size_t)nstrip * NT_PER_STRIP * 16 * H2;

    // prologue: fetch tile 0 into registers (each thread owns 32B of the tile)
    v8h g0 = *(const v8h*)(bstrip + tid * 8);
    v8h g1 = *(const v8h*)(bstrip + 2048 + tid * 8);

    #pragma unroll 1
    for (int nt = 0; nt < NT_PER_STRIP; ++nt) {
        int buf = nt & 1;
        // commit current tile to LDS
        *(v8h*)&bsh[buf][tid * 8]        = g0;
        *(v8h*)&bsh[buf][2048 + tid * 8] = g1;
        // start fetching the next tile (overlaps with this tile's WMMAs)
        if (nt + 1 < NT_PER_STRIP) {
            const _Float16* bn = bstrip + (size_t)(nt + 1) * (16 * H2);
            g0 = *(const v8h*)(bn + tid * 8);
            g1 = *(const v8h*)(bn + 2048 + tid * 8);
        }
        __syncthreads();

        // B fragments out of LDS (same 16-bit 16x32 layout, row-major tile)
        const _Float16* bl = &bsh[buf][m16 * H2 + hs * 8];
        v8f c = {};
        #pragma unroll
        for (int kc = 0; kc < 8; ++kc) {
            union { v16h v; v8h h[2]; } ub;
            ub.h[0] = *(const v8h*)(bl + kc * 32);
            ub.h[1] = *(const v8h*)(bl + kc * 32 + 16);
            c = __builtin_amdgcn_wmma_f32_16x16x32_f16(
                    false, a[kc], false, ub.v, (short)0, c, false, false);
        }

        int ntile = nstrip * NT_PER_STRIP + nt;
        float bi = bias[ntile * 16 + m16];
        // C/D layout: lane -> N = lane&15; VGPR r -> M = r + (lane>=16 ? 8 : 0)
        float* yp = Y + ((size_t)(mtile * 16 + hs * 8)) * NLOC + ntile * 16 + m16;
        #pragma unroll
        for (int r = 0; r < 8; ++r)
            __builtin_nontemporal_store(c[r] + bi, yp + (size_t)r * NLOC);
    }
}

// ---------------------------------------------------------------------------
// launch
// ---------------------------------------------------------------------------
extern "C" void kernel_launch(void* const* d_in, const int* in_sizes, int n_in,
                              void* d_out, int out_size, void* d_ws, size_t ws_size,
                              hipStream_t stream) {
    const int*   x    = (const int*)  d_in[0];
    const float* t    = (const float*)d_in[1];
    const float* sp   = (const float*)d_in[2];
    const float* h0   = (const float*)d_in[5];
    const int*   au   = (const int*)  d_in[6];
    const float* enc  = (const float*)d_in[7];
    const float* uemb = (const float*)d_in[8];
    const float* pref = (const float*)d_in[9];
    const float* W_ih = (const float*)d_in[10];
    const float* b_ih = (const float*)d_in[11];
    const float* W_hh = (const float*)d_in[12];
    const float* b_hh = (const float*)d_in[13];
    const float* Wp   = (const float*)d_in[14];
    const float* bp   = (const float*)d_in[15];
    const float* W_fc = (const float*)d_in[16];
    const float* b_fc = (const float*)d_in[17];

    char* ws = (char*)d_ws;
    float*    xemb = (float*)(ws + OFF_XEMB);
    float*    outr = (float*)(ws + OFF_OUT);
    float*    pu   = (float*)(ws + OFF_PU);
    float*    sim  = (float*)(ws + OFF_SIM);
    float*    coef = (float*)(ws + OFF_COEF);
    _Float16* Af   = (_Float16*)(ws + OFF_A);
    _Float16* Wh   = (_Float16*)(ws + OFF_WH);

    float* y    = (float*)d_out;                  // (S,U,NLOC)
    float* hout = y + (size_t)M * NLOC;           // (1,U,H)

    k_gather_x<<<(S * U * H + 255) / 256, 256, 0, stream>>>(x, enc, xemb);
    k_gather_u<<<(U * H + 255) / 256, 256, 0, stream>>>(au, uemb, pu);
    k_rnn<<<U, H, 0, stream>>>(xemb, h0, W_ih, b_ih, W_hh, b_hh, outr, hout);
    k_sim<<<U, H2, 0, stream>>>(pu, xemb, Wp, bp, pref, sim);
    k_coef<<<(S * U + 255) / 256, 256, 0, stream>>>(t, sp, sim, coef);
    k_pack_a<<<(M * H2 + 255) / 256, 256, 0, stream>>>(outr, coef, pu, Af);
    k_pack_w<<<(NLOC * H2 + 255) / 256, 256, 0, stream>>>(W_fc, Wh);

    k_gemm<<<MG * NSTRIPS, 256, 0, stream>>>(Af, Wh, b_fc, y);   // 3200 WGs
}
